// FeatureExtractor_86500641341819
// MI455X (gfx1250) — matrix operand order repaired and checked by reading, not compiled, not run on hardware
//
#include <hip/hip_runtime.h>
#include <hip/hip_bf16.h>

typedef __attribute__((ext_vector_type(16))) _Float16 v16h;
typedef __attribute__((ext_vector_type(8)))  float    v8f;

#define GB    64
#define GN    1024
#define GDEG  8
#define GFIN  14
#define NHID  128
#define GE    (GB * GN * GDEG)     // 524288
#define K1V   820
#define K2V   656
#define NEG_INF (-__builtin_inff())

// ---------------------------------------------------------------------------
// Edge aggregation, stage 1: agg[dst][0..13] += x[src][0..13]  (emask all 1)
// ---------------------------------------------------------------------------
__global__ void agg1_kernel(const float* __restrict__ x,
                            const int* __restrict__ src,
                            const int* __restrict__ dst,
                            float* __restrict__ agg, int nE) {
    int e = blockIdx.x * blockDim.x + threadIdx.x;
    if (e >= nE) return;
    int s = src[e], d = dst[e];
    const float* xs = x + (size_t)s * GFIN;
    float* ad = agg + (size_t)d * GFIN;
#pragma unroll
    for (int f = 0; f < GFIN; ++f) atomicAdd(&ad[f], xs[f]);
}

// ---------------------------------------------------------------------------
// Edge aggregation, stage 2: masked 128-dim scatter sum.
// thread = (edge, 4-float chunk); 32 chunks per edge.
// ---------------------------------------------------------------------------
__global__ void agg2_kernel(const float* __restrict__ h,
                            const int* __restrict__ src2,
                            const int* __restrict__ dst2,
                            const int* __restrict__ em2,
                            float* __restrict__ agg, int nE) {
    int t = blockIdx.x * blockDim.x + threadIdx.x;
    int e = t >> 5;
    if (e >= nE) return;
    if (!em2[e]) return;
    int c = (t & 31) * 4;
    const float* hs = h + (size_t)src2[e] * NHID + c;
    float* ad = agg + (size_t)dst2[e] * NHID + c;
#pragma unroll
    for (int j = 0; j < 4; ++j) atomicAdd(&ad[j], hs[j]);
}

// ---------------------------------------------------------------------------
// Conv1 fused GEMM (K = 14+14 -> one 16x16x32 WMMA chunk, zero padded).
// One wave -> one 16x16 tile; block = 8 waves = one M-row x all 8 N-tiles.
// ---------------------------------------------------------------------------
__device__ __forceinline__ float fetchA1(const float* agg, const float* x,
                                         int m, int k) {
    if (k < GFIN)       return agg[(size_t)m * GFIN + k];
    k -= GFIN;
    if (k < GFIN)       return x[(size_t)m * GFIN + k];
    return 0.0f;
}

__device__ __forceinline__ float fetchB1(const float* Wrel, const float* Wroot,
                                         int n, int k) {
    if (k < GFIN)       return Wrel[(size_t)n * GFIN + k];
    k -= GFIN;
    if (k < GFIN)       return Wroot[(size_t)n * GFIN + k];
    return 0.0f;
}

__global__ __launch_bounds__(256)
void conv1_gemm_kernel(const float* __restrict__ agg,
                       const float* __restrict__ x,
                       const float* __restrict__ Wrel,
                       const float* __restrict__ Wroot,
                       const float* __restrict__ bias,
                       float* __restrict__ out) {
    const int wave = threadIdx.x >> 5;         // 0..7 -> N tile
    const int lane = threadIdx.x & 31;
    const int m0   = blockIdx.x * 16;
    const int n0   = wave * 16;
    const int mrow = lane & 15;
    const int ncol = lane & 15;
    const int kh   = lane >> 4;

    v16h af, bf;
#pragma unroll
    for (int j = 0; j < 8; ++j) {
        int kbA = (j < 4 ? 2 * j : 16 + 2 * (j - 4)) + 8 * kh;
        af[2 * j]     = (_Float16)fetchA1(agg, x, m0 + mrow, kbA);
        af[2 * j + 1] = (_Float16)fetchA1(agg, x, m0 + mrow, kbA + 1);
        int kbB = (j < 4 ? 2 * j : 16 + 2 * (j - 4)) + 8 * kh;
        bf[2 * j]     = (_Float16)fetchB1(Wrel, Wroot, n0 + ncol, kbB);
        bf[2 * j + 1] = (_Float16)fetchB1(Wrel, Wroot, n0 + ncol, kbB + 1);
    }
    v8f acc = {};
    acc = __builtin_amdgcn_wmma_f32_16x16x32_f16(false, af, false, bf,
                                                 (short)0, acc, false, false);
    const int nout = n0 + ncol;
    const float bn = bias[nout];
#pragma unroll
    for (int r = 0; r < 8; ++r) {
        int m = m0 + r + 8 * kh;
        float v = acc[r] + bn;
        out[(size_t)m * NHID + nout] = v > 0.0f ? v : 0.0f;
    }
}

// ---------------------------------------------------------------------------
// Conv2 fused GEMM, K = 128+128 = 256 (8 WMMA chunks).
//  - B fragments (weights) preloaded once per wave into registers,
//    amortized over a grid-stride loop of M tiles.
//  - A tile (16 x 256 f32 = 16KB) staged to LDS with
//    GLOBAL_LOAD_ASYNC_TO_LDS_B128 (ASYNCcnt path), consumed via ds loads.
// ---------------------------------------------------------------------------
#define TPAD 260   // 256 + 4 floats row padding (keeps rows 16B-aligned)

__global__ __launch_bounds__(256)
void conv2_gemm_kernel(const float* __restrict__ agg,
                       const float* __restrict__ x,
                       const float* __restrict__ Wrel,
                       const float* __restrict__ Wroot,
                       const float* __restrict__ bias,
                       float* __restrict__ out, int Mtiles) {
    __shared__ float tileA[16][TPAD];

    const int wave = threadIdx.x >> 5;         // 0..7 -> N tile
    const int lane = threadIdx.x & 31;
    const int n0   = wave * 16;
    const int mrow = lane & 15;
    const int ncol = lane & 15;
    const int kh   = lane >> 4;

    // ---- preload all B fragments (8 K-chunks) into registers ----
    v16h bfr[8];
#pragma unroll
    for (int kc = 0; kc < 8; ++kc) {
#pragma unroll
        for (int j = 0; j < 8; ++j) {
            int kbB = kc * 32 + (j < 4 ? 2 * j : 16 + 2 * (j - 4)) + 8 * kh;     // even, pair stays in half
            float b0, b1;
            if (kbB < NHID) {
                b0 = Wrel[(size_t)(n0 + ncol) * NHID + kbB];
                b1 = Wrel[(size_t)(n0 + ncol) * NHID + kbB + 1];
            } else {
                b0 = Wroot[(size_t)(n0 + ncol) * NHID + kbB - NHID];
                b1 = Wroot[(size_t)(n0 + ncol) * NHID + kbB - NHID + 1];
            }
            bfr[kc][2 * j]     = (_Float16)b0;
            bfr[kc][2 * j + 1] = (_Float16)b1;
        }
    }

    // LDS byte offset of the staging tile (generic LDS ptr low 32 bits).
    const unsigned tileBase = (unsigned)(uintptr_t)(&tileA[0][0]);
    const int nout = n0 + ncol;
    const float bn = bias[nout];

    for (int mt = blockIdx.x; mt < Mtiles; mt += gridDim.x) {
        const int m0 = mt * 16;

        // ---- async-stage A tile: 16 rows x 256 cols f32 (agg || x) ----
#pragma unroll
        for (int j = 0; j < 4; ++j) {
            int f4  = threadIdx.x + 256 * j;   // 1024 float4s total
            int r   = f4 >> 6;                 // 64 float4 per row
            int c4  = f4 & 63;
            int col = c4 * 4;
            const float* gsrc = (col < NHID)
                ? (agg + (size_t)(m0 + r) * NHID + col)
                : (x   + (size_t)(m0 + r) * NHID + (col - NHID));
            unsigned ldsOff = tileBase + (unsigned)(r * (TPAD * 4) + c4 * 16);
            asm volatile("global_load_async_to_lds_b128 %0, %1, off"
                         :: "v"(ldsOff), "v"(gsrc) : "memory");
        }
        asm volatile("s_wait_asynccnt 0" ::: "memory");
        __syncthreads();

        // ---- 8 WMMA chunks over K = 256 ----
        v8f acc = {};
        const float* arow = &tileA[mrow][0];
#pragma unroll
        for (int kc = 0; kc < 8; ++kc) {
            v16h af;
#pragma unroll
            for (int j = 0; j < 8; ++j) {
                int kbA = kc * 32 + (j < 4 ? 2 * j : 16 + 2 * (j - 4)) + 8 * kh;
                af[2 * j]     = (_Float16)arow[kbA];
                af[2 * j + 1] = (_Float16)arow[kbA + 1];
            }
            acc = __builtin_amdgcn_wmma_f32_16x16x32_f16(
                false, af, false, bfr[kc], (short)0, acc, false, false);
        }

        // ---- write D with fused bias + relu ----
#pragma unroll
        for (int r = 0; r < 8; ++r) {
            int m = m0 + r + 8 * kh;
            float v = acc[r] + bn;
            out[(size_t)m * NHID + nout] = v > 0.0f ? v : 0.0f;
        }
        __syncthreads();   // tile consumed before next iteration overwrites
    }
}

// ---------------------------------------------------------------------------
// score[i] = tanh( h[i]·w / ||w|| ); one wave per node.
// ---------------------------------------------------------------------------
__global__ void score_kernel(const float* __restrict__ h,
                             const float* __restrict__ w,
                             float* __restrict__ score, int nnodes) {
    int wid  = (blockIdx.x * blockDim.x + threadIdx.x) >> 5;
    int lane = threadIdx.x & 31;
    if (wid >= nnodes) return;
    float dot = 0.0f, nrm2 = 0.0f;
    const float* hr = h + (size_t)wid * NHID;
#pragma unroll
    for (int j = lane; j < NHID; j += 32) {
        float wv = w[j];
        nrm2 += wv * wv;
        dot  += hr[j] * wv;
    }
#pragma unroll
    for (int off = 16; off > 0; off >>= 1) {
        dot  += __shfl_down(dot, off, 32);
        nrm2 += __shfl_down(nrm2, off, 32);
    }
    if (lane == 0) score[wid] = tanhf(dot * rsqrtf(nrm2));
}

// ---------------------------------------------------------------------------
// Per-graph top-k: bitonic sort of 1024 (score, idx) pairs in LDS, descending,
// stable (ties -> lower index first) to match jax.lax.top_k.
// ---------------------------------------------------------------------------
__global__ __launch_bounds__(1024)
void topk_kernel(const float* __restrict__ score,
                 float* __restrict__ vals, int* __restrict__ perm,
                 int* __restrict__ newpos, int* __restrict__ kept,
                 int n_per, int k) {
    __shared__ float sv[1024];
    __shared__ int   si[1024];
    const int g = blockIdx.x;
    const int t = threadIdx.x;
    sv[t] = (t < n_per) ? score[(size_t)g * n_per + t] : NEG_INF;
    si[t] = t;
    for (unsigned size = 2; size <= 1024; size <<= 1) {
        for (unsigned stride = size >> 1; stride > 0; stride >>= 1) {
            __syncthreads();
            unsigned i = t, p = t ^ stride;
            if (p > i) {
                bool up = ((i & size) == 0);            // descending block
                float v1 = sv[i], v2 = sv[p];
                int   i1 = si[i], i2 = si[p];
                bool lt = (v1 < v2) || (v1 == v2 && i1 > i2);
                if (lt == up) { sv[i] = v2; sv[p] = v1; si[i] = i2; si[p] = i1; }
            }
        }
    }
    __syncthreads();
    if (t < k) {
        int pold = g * n_per + si[t];
        int pnew = g * k + t;
        perm[pnew]   = pold;
        vals[pnew]   = sv[t];
        newpos[pold] = pnew;
        kept[pold]   = 1;
    }
}

// ---------------------------------------------------------------------------
// new_x[r][f] = h[perm[r]][f] * vals[r]
// ---------------------------------------------------------------------------
__global__ void gather_scale_kernel(const float* __restrict__ h,
                                    const int* __restrict__ perm,
                                    const float* __restrict__ vals,
                                    float* __restrict__ out, int rows) {
    int idx = blockIdx.x * blockDim.x + threadIdx.x;
    if (idx >= rows * NHID) return;
    int r = idx >> 7, f = idx & (NHID - 1);
    out[idx] = h[(size_t)perm[r] * NHID + f] * vals[r];
}

// ---------------------------------------------------------------------------
// Remap edges after pool 1.
// ---------------------------------------------------------------------------
__global__ void remap_edges_kernel(const int* __restrict__ src,
                                   const int* __restrict__ dst,
                                   const int* __restrict__ newpos,
                                   const int* __restrict__ kept,
                                   int* __restrict__ src2, int* __restrict__ dst2,
                                   int* __restrict__ em2, int nE) {
    int e = blockIdx.x * blockDim.x + threadIdx.x;
    if (e >= nE) return;
    int s = src[e], d = dst[e];
    em2[e]  = kept[s] & kept[d];
    src2[e] = newpos[s];
    dst2[e] = newpos[d];
}

// ---------------------------------------------------------------------------
// Readout: [B, k, 128] -> mean/max over node axis, concat -> [B, 256].
// ---------------------------------------------------------------------------
__global__ void readout_kernel(const float* __restrict__ xp,
                               float* __restrict__ out, int k, int accum) {
    int g = blockIdx.x, f = threadIdx.x;
    const float* base = xp + (size_t)g * k * NHID + f;
    float sm = 0.0f, mx = NEG_INF;
    for (int i = 0; i < k; ++i) {
        float v = base[(size_t)i * NHID];
        sm += v;
        mx = fmaxf(mx, v);
    }
    float mean = sm / (float)k;
    if (accum) {
        out[g * 256 + f]        += mean;
        out[g * 256 + NHID + f] += mx;
    } else {
        out[g * 256 + f]        = mean;
        out[g * 256 + NHID + f] = mx;
    }
}

// ---------------------------------------------------------------------------
extern "C" void kernel_launch(void* const* d_in, const int* in_sizes, int n_in,
                              void* d_out, int out_size, void* d_ws, size_t ws_size,
                              hipStream_t stream) {
    const float* x      = (const float*)d_in[0];
    const int*   eix    = (const int*)d_in[1];          // [2, E]
    const int    nE     = in_sizes[1] / 2;              // 524288
    const int*   src    = eix;
    const int*   dst    = eix + nE;
    const float* W1rel  = (const float*)d_in[3];
    const float* b1rel  = (const float*)d_in[4];
    const float* W1root = (const float*)d_in[5];
    const float* p1w    = (const float*)d_in[6];
    const float* W2rel  = (const float*)d_in[7];
    const float* b2rel  = (const float*)d_in[8];
    const float* W2root = (const float*)d_in[9];
    const float* p2w    = (const float*)d_in[10];
    float*       out    = (float*)d_out;

    const int Nnodes = GB * GN;        // 65536
    const int M1     = Nnodes;         // conv1 rows
    const int M2     = GB * K1V;       // 52480 conv2 rows
    const int R2     = GB * K2V;       // 41984 pooled-2 rows

    // ---- workspace carve-out ----
    float* agg1 = (float*)d_ws;                    // 65536*14
    float* h1   = agg1 + (size_t)Nnodes * GFIN;    // 65536*128  (reused as h2)
    float* h1p  = h1   + (size_t)Nnodes * NHID;    // 52480*128
    float* agg2 = h1p  + (size_t)M2 * NHID;        // 52480*128
    float* h2p  = agg2 + (size_t)M2 * NHID;        // 41984*128
    float* scr  = h2p  + (size_t)R2 * NHID;        // 65536
    float* vals = scr  + Nnodes;                   // 52480
    int* perm   = (int*)(vals + M2);               // 52480
    int* newpos = perm + M2;                       // 65536
    int* kept   = newpos + Nnodes;                 // 65536
    int* src2   = kept + Nnodes;                   // E
    int* dst2   = src2 + nE;                       // E
    int* em2    = dst2 + nE;                       // E
    float* h2   = h1;                              // alias: h1 dead after gather

    // ================= Stage 1: GraphConv1 + pool1 + readout1 ==============
    hipMemsetAsync(agg1, 0, (size_t)Nnodes * GFIN * sizeof(float), stream);
    agg1_kernel<<<(nE + 255) / 256, 256, 0, stream>>>(x, src, dst, agg1, nE);

    conv1_gemm_kernel<<<M1 / 16, 256, 0, stream>>>(
        agg1, x, W1rel, W1root, b1rel, h1);

    score_kernel<<<(Nnodes * 32 + 255) / 256, 256, 0, stream>>>(h1, p1w, scr, Nnodes);

    hipMemsetAsync(kept, 0, (size_t)Nnodes * sizeof(int), stream);
    hipMemsetAsync(newpos, 0, (size_t)Nnodes * sizeof(int), stream);
    topk_kernel<<<GB, 1024, 0, stream>>>(scr, vals, perm, newpos, kept, GN, K1V);

    gather_scale_kernel<<<(M2 * NHID + 255) / 256, 256, 0, stream>>>(
        h1, perm, vals, h1p, M2);

    readout_kernel<<<GB, NHID, 0, stream>>>(h1p, out, K1V, /*accum=*/0);

    remap_edges_kernel<<<(nE + 255) / 256, 256, 0, stream>>>(
        src, dst, newpos, kept, src2, dst2, em2, nE);

    // ================= Stage 2: GraphConv2 + pool2 + readout2 ==============
    hipMemsetAsync(agg2, 0, (size_t)M2 * NHID * sizeof(float), stream);
    agg2_kernel<<<((size_t)nE * 32 + 255) / 256, 256, 0, stream>>>(
        h1p, src2, dst2, em2, agg2, nE);

    conv2_gemm_kernel<<<M2 / 16 / 4, 256, 0, stream>>>(
        agg2, h1p, W2rel, W2root, b2rel, h2, M2 / 16);

    score_kernel<<<(M2 * 32 + 255) / 256, 256, 0, stream>>>(h2, p2w, scr, M2);

    hipMemsetAsync(kept, 0, (size_t)M2 * sizeof(int), stream);
    topk_kernel<<<GB, 1024, 0, stream>>>(scr, vals, perm, newpos, kept, K1V, K2V);

    gather_scale_kernel<<<(R2 * NHID + 255) / 256, 256, 0, stream>>>(
        h2, perm, vals, h2p, R2);

    readout_kernel<<<GB, NHID, 0, stream>>>(h2p, out, K2V, /*accum=*/1);
}
